// PrunedCaps_54365696033051
// MI455X (gfx1250) — matrix operand order repaired and checked by reading, not compiled, not run on hardware
//
#include <hip/hip_runtime.h>

// ---------------------------------------------------------------------------
// PrunedCaps for MI455X (gfx1250, wave32, WMMA).
// u: [B=16, n=2048, D=128] f32.  Gram matrix of normalized rows via
// v_wmma_f32_16x16x32_f16 (f16 in, f32 accum), upper-triangle threshold at
// theta=0.7, stable compaction of kept rows, zero-padded tail.
// ---------------------------------------------------------------------------

typedef __attribute__((ext_vector_type(16))) _Float16 v16h;
typedef __attribute__((ext_vector_type(8)))  _Float16 v8h;
typedef __attribute__((ext_vector_type(4)))  _Float16 v4h;
typedef __attribute__((ext_vector_type(8)))  float    v8f;

#define PC_B   16
#define PC_N   2048
#define PC_D   128
#define PC_NT  128          // PC_N / 16 tiles
#define PC_THETA 0.7f
#define PC_EPS   1e-8f

// ---- 1) L2 norms: one wave per vector, float4 per lane --------------------
__global__ void pc_norms_kernel(const float* __restrict__ u,
                                float* __restrict__ norms) {
  int vec  = (int)((blockIdx.x * blockDim.x + threadIdx.x) >> 5); // [0, B*N)
  int lane = threadIdx.x & 31;
  float4 v = *(const float4*)(u + (size_t)vec * PC_D + lane * 4);
  float s = v.x * v.x + v.y * v.y + v.z * v.z + v.w * v.w;
#pragma unroll
  for (int off = 16; off > 0; off >>= 1) s += __shfl_xor(s, off, 32);
  if (lane == 0) norms[vec] = sqrtf(s);
}

// ---- 2) per-batch bitonic argsort ascending (n=2048, 1 block) -------------
__global__ void pc_sort_kernel(const float* __restrict__ norms,
                               int* __restrict__ sidx,
                               float* __restrict__ snorm) {
  __shared__ float key[PC_N];
  __shared__ int   val[PC_N];
  const int n = PC_N;
  int b = blockIdx.x;
  for (int i = threadIdx.x; i < n; i += blockDim.x) {
    key[i] = norms[b * n + i];
    val[i] = i;
  }
  __syncthreads();
  for (int k = 2; k <= n; k <<= 1) {
    for (int j = k >> 1; j > 0; j >>= 1) {
      for (int i = threadIdx.x; i < n; i += blockDim.x) {
        int ixj = i ^ j;
        if (ixj > i) {
          bool asc = ((i & k) == 0);
          float ki = key[i], kj = key[ixj];
          if (asc ? (ki > kj) : (ki < kj)) {
            key[i] = kj; key[ixj] = ki;
            int t = val[i]; val[i] = val[ixj]; val[ixj] = t;
          }
        }
      }
      __syncthreads();
    }
  }
  for (int i = threadIdx.x; i < n; i += blockDim.x) {
    sidx[b * n + i]  = val[i];
    snorm[b * n + i] = key[i];
  }
}

// ---- 3) gather ordered rows, normalize, convert to f16 --------------------
__global__ void pc_gather_norm_kernel(const float* __restrict__ u,
                                      const int* __restrict__ sidx,
                                      const float* __restrict__ snorm,
                                      _Float16* __restrict__ un) {
  int vec  = (int)((blockIdx.x * blockDim.x + threadIdx.x) >> 5); // [0, B*N)
  int lane = threadIdx.x & 31;
  int b    = vec >> 11;                 // / PC_N
  int src  = sidx[vec];                 // index within batch
  float inv = 1.0f / (snorm[vec] + PC_EPS);
  float4 v = *(const float4*)(u + (((size_t)b << 11) + src) * PC_D + lane * 4);
  v4h h;
  h[0] = (_Float16)(v.x * inv);
  h[1] = (_Float16)(v.y * inv);
  h[2] = (_Float16)(v.z * inv);
  h[3] = (_Float16)(v.w * inv);
  *(v4h*)(un + (size_t)vec * PC_D + lane * 4) = h;
}

// ---- 4) WMMA prune --------------------------------------------------------
// A 16x32 f16 layout: lane L<16 holds row L, K k0..k0+15;
//                     lane L>=16 holds row L-16, K k0+16..k0+31 (contiguous k, same order as B).
// B 32x16 f16 layout: lane L holds col L%16, 16 consecutive K from k0+(L/16)*16.
// C f32 layout: VGPR v -> row v + 8*(L>=16); col = L%16.

// One 16x16 cos tile: rows from the preloaded A frags, cols jT*16..+15.
__device__ __forceinline__ v8f pc_cos_tile(const _Float16* __restrict__ bbase,
                                           const v16h afrag[4]) {
  v8f c = {};
#pragma unroll
  for (int kk = 0; kk < 4; ++kk) {
    v16h bfrag = *(const v16h*)(bbase + kk * 32);
    c = __builtin_amdgcn_wmma_f32_16x16x32_f16(
        /*neg_a=*/false, afrag[kk], /*neg_b=*/false, bfrag,
        /*c_mod=*/(short)0, c, /*reuse_a=*/false, /*reuse_b=*/false);
  }
  return c;
}

__global__ void __launch_bounds__(256)
pc_prune_kernel(const _Float16* __restrict__ un,
                unsigned* __restrict__ pruned) {
  int b    = blockIdx.x >> 7;       // / PC_NT
  int iT   = blockIdx.x & (PC_NT - 1);
  // Scalarize the wave id so the j-loop is a pure SALU loop (EXEC untouched,
  // WMMA always issued with EXEC == all ones, no per-iteration v_cmp).
  int wave = __builtin_amdgcn_readfirstlane((int)(threadIdx.x >> 5));
  int lane = threadIdx.x & 31;
  int m    = lane & 15;
  int half = lane >> 4;

  __shared__ unsigned rowFlags;
  if (threadIdx.x == 0) rowFlags = 0u;
  __syncthreads();

  // Preload the A fragments (rows iT*16 .. +15, K = 0..127 as 4 slices).
  const _Float16* abase = un + (((size_t)b * PC_N) + iT * 16 + m) * PC_D;
  v16h afrag[4];
#pragma unroll
  for (int kk = 0; kk < 4; ++kk) {
    const _Float16* p = abase + kk * 32 + half * 16;
    v8h lo = *(const v8h*)(p);
    v8h hi = *(const v8h*)(p + 8);
    afrag[kk] = __builtin_shufflevector(lo, hi,
        0, 1, 2, 3, 4, 5, 6, 7, 8, 9, 10, 11, 12, 13, 14, 15);
  }

  const _Float16* ubb = un + ((size_t)b * PC_N) * PC_D;
  unsigned bits = 0u;

  // Peeled diagonal tile (wave 0 only; scalar branch): needs the j>i mask.
  if (wave == 0) {
    v8f c = pc_cos_tile(ubb + ((size_t)iT * 16 + m) * PC_D + half * 16, afrag);
    int gj  = iT * 16 + m;          // this lane's column (capsule j)
    int gib = iT * 16 + half * 8;   // first row this lane holds
#pragma unroll
    for (int v = 0; v < 8; ++v)
      if (c[v] > PC_THETA && gj > gib + v) bits |= 1u << (half * 8 + v);
  }

  // Strict upper tiles: every (row, col) is valid -> just max-accumulate.
  v8f maxc;
#pragma unroll
  for (int v = 0; v < 8; ++v) maxc[v] = -3.0e38f;

  for (int jT = iT + 1 + wave; jT < PC_NT; jT += 8) {
    v8f c = pc_cos_tile(ubb + ((size_t)jT * 16 + m) * PC_D + half * 16, afrag);
#pragma unroll
    for (int v = 0; v < 8; ++v) maxc[v] = fmaxf(maxc[v], c[v]);
  }
#pragma unroll
  for (int v = 0; v < 8; ++v)
    if (maxc[v] > PC_THETA) bits |= 1u << (half * 8 + v);

  // OR-reduce across the wave (bit positions are disjoint per row).
#pragma unroll
  for (int off = 16; off > 0; off >>= 1) bits |= __shfl_xor(bits, off, 32);
  if (lane == 0 && bits) atomicOr(&rowFlags, bits);
  __syncthreads();
  if (threadIdx.x < 16)
    pruned[((size_t)b * PC_N) + iT * 16 + threadIdx.x] =
        (rowFlags >> threadIdx.x) & 1u;
}

// ---- 5) per-batch stable compaction scan ----------------------------------
__global__ void pc_compact_kernel(const unsigned* __restrict__ pruned,
                                  int* __restrict__ keptIdx,
                                  int* __restrict__ count) {
  __shared__ int partial[256];
  int b = blockIdx.x, tid = threadIdx.x;
  int base = b * PC_N + tid * 8;
  int keep[8], loc[8], s = 0;
#pragma unroll
  for (int e = 0; e < 8; ++e) {
    keep[e] = (pruned[base + e] == 0u);
    loc[e] = s;
    s += keep[e];
  }
  partial[tid] = s;
  __syncthreads();
  for (int off = 1; off < 256; off <<= 1) {
    int v = (tid >= off) ? partial[tid - off] : 0;
    __syncthreads();
    partial[tid] += v;
    __syncthreads();
  }
  int offset = (tid > 0) ? partial[tid - 1] : 0;
#pragma unroll
  for (int e = 0; e < 8; ++e)
    if (keep[e]) keptIdx[b * PC_N + offset + loc[e]] = tid * 8 + e;
  if (tid == 255) count[b] = partial[255];
}

// ---- 6) output gather (compose indices), zero-fill tail -------------------
__global__ void pc_output_kernel(const float* __restrict__ u,
                                 const int* __restrict__ sidx,
                                 const int* __restrict__ keptIdx,
                                 const int* __restrict__ count,
                                 float* __restrict__ out) {
  int row  = blockIdx.x;            // [0, B*N)
  int b    = row >> 11;
  int r    = row & (PC_N - 1);
  int lane = threadIdx.x;           // 32 threads
  float4 val = make_float4(0.0f, 0.0f, 0.0f, 0.0f);
  if (r < count[b]) {
    int src = sidx[b * PC_N + keptIdx[b * PC_N + r]];
    val = *(const float4*)(u + (((size_t)b << 11) + src) * PC_D + lane * 4);
  }
  *(float4*)(out + (size_t)row * PC_D + lane * 4) = val;
}

// ---------------------------------------------------------------------------
extern "C" void kernel_launch(void* const* d_in, const int* in_sizes, int n_in,
                              void* d_out, int out_size, void* d_ws,
                              size_t ws_size, hipStream_t stream) {
  (void)in_sizes; (void)n_in; (void)out_size; (void)ws_size;
  const float* u = (const float*)d_in[0];
  float* out = (float*)d_out;

  // Workspace layout (256B-aligned slabs, ~9 MB total):
  char* ws = (char*)d_ws;
  float*    norms   = (float*)(ws + 0);          // 32768 f32
  int*      sidx    = (int*)(ws + 131072);       // 32768 i32
  float*    snorm   = (float*)(ws + 262144);     // 32768 f32
  unsigned* pruned  = (unsigned*)(ws + 393216);  // 32768 u32 (fully rewritten)
  int*      keptIdx = (int*)(ws + 524288);       // 32768 i32
  int*      count   = (int*)(ws + 655360);       // 16 i32
  _Float16* un      = (_Float16*)(ws + 655616);  // 32768*128 f16 = 8 MB

  const int BN = PC_B * PC_N;                    // 32768 vectors

  pc_norms_kernel<<<BN / 8, 256, 0, stream>>>(u, norms);
  pc_sort_kernel<<<PC_B, 1024, 0, stream>>>(norms, sidx, snorm);
  pc_gather_norm_kernel<<<BN / 8, 256, 0, stream>>>(u, sidx, snorm, un);
  pc_prune_kernel<<<PC_B * PC_NT, 256, 0, stream>>>(un, pruned);
  pc_compact_kernel<<<PC_B, 256, 0, stream>>>(pruned, keptIdx, count);
  pc_output_kernel<<<BN, 32, 0, stream>>>(u, sidx, keptIdx, count, out);
}
